// PNet2_34308198761178
// MI455X (gfx1250) — compile-verified, matrix operand order checked by reading, not run on hardware
//
#include <hip/hip_runtime.h>
#include <hip/hip_bf16.h>
#include <math.h>

typedef __attribute__((ext_vector_type(16))) _Float16 v16h;
typedef __attribute__((ext_vector_type(8)))  _Float16 v8h;
typedef __attribute__((ext_vector_type(8)))  float    v8f;

#define NW 8   // waves per 256-thread block (wave32)

// ---- A fragment from LDS (CDNA5 16-bit A 16x32 layout): two aligned 16B chunks ----
// lane l: m = l&15, kh = l>>4 ; halfs [kh*8 .. kh*8+7] and [16+kh*8 .. 16+kh*8+7]
__device__ __forceinline__ v16h load_a_lds(const _Float16* A, int ldm, int k0, int m16, int kh) {
  const _Float16* p = A + m16 * ldm + k0 + (kh << 3);
  v8h lo = *(const v8h*)p;
  v8h hi = *(const v8h*)(p + 16);
  return __builtin_shufflevector(lo, hi, 0, 1, 2, 3, 4, 5, 6, 7, 8, 9, 10, 11, 12, 13, 14, 15);
}

// ---- B fragment from pre-swizzled weights: 32 contiguous bytes per lane ----
// layout: [((t*S + si)*32 + lane) * 16] halfs, fully coalesced across the wave
__device__ __forceinline__ v16h load_b_sw(const _Float16* W, int S, int t, int si, int lane) {
  const v8h* p = (const v8h*)(W + (((size_t)(t * S + si) << 5) + lane) * 16);
  v8h lo = p[0];
  v8h hi = p[1];
  return __builtin_shufflevector(lo, hi, 0, 1, 2, 3, 4, 5, 6, 7, 8, 9, 10, 11, 12, 13, 14, 15);
}

// ---------------------------------------------------------------------------
// Fused grouped PointConv MLP: per query, stage 16-neighbor feature blocks in
// LDS, run 3 dense layers with v_wmma_f32_16x16x32_f16, max-aggregate over K.
// mode 0: feat = rel(pos[nidx]-qpos)            (Cx=0)
// mode 1: feat = [x[nidx] , rel]                (gathered)
// mode 2: feat = [x[row]  , pos[row]]           (direct rows, no gather)
// ---------------------------------------------------------------------------
__global__ void __launch_bounds__(256)
grouped_mlp_kernel(int mode, int K, int Cx, int xStride,
                   const _Float16* __restrict__ xin,
                   const float* __restrict__ posAll,
                   const float* __restrict__ qposArr,
                   const int* __restrict__ nidx,
                   int CinP, int C1, int C2, int C3P,
                   const _Float16* __restrict__ W1, const float* __restrict__ B1,
                   const _Float16* __restrict__ W2, const float* __restrict__ B2,
                   const _Float16* __restrict__ W3, const float* __restrict__ B3,
                   _Float16* __restrict__ outF16, float* __restrict__ outF32,
                   int outStride, int outOff, int doMax)
{
  __shared__ __align__(16) char smem[57600];          // max: head 16*(1024+512+256)*2 = 57344 B
  _Float16* feat = (_Float16*)smem;                   // [16, CinP]
  _Float16* h1   = feat + 16 * CinP;                  // [16, C1]
  _Float16* h2   = h1 + 16 * C1;                      // [16, C2]

  const int tid  = threadIdx.x;
  const int wave = tid >> 5;
  const int lane = tid & 31;
  const int col  = lane & 15;
  const int kh   = lane >> 4;
  const int q    = blockIdx.x;

  float qx = 0.f, qy = 0.f, qz = 0.f;
  if (mode < 2) { qx = qposArr[q * 3 + 0]; qy = qposArr[q * 3 + 1]; qz = qposArr[q * 3 + 2]; }

  float accum[8];
#pragma unroll
  for (int i = 0; i < 8; ++i) accum[i] = -3.0e38f;

  const int tiles1 = C1 >> 4, tiles2 = C2 >> 4, tiles3 = C3P >> 4;
  const int S1 = CinP >> 5, S2 = C1 >> 5, S3 = C2 >> 5;
  const int nblk = K >> 4;

  for (int blk = 0; blk < nblk; ++blk) {
    // ---- stage 16 rows of input features (f16, zero padded); 2 rows per wave ----
    for (int r = wave * 2; r < wave * 2 + 2; ++r) {
      int rr = (blk << 4) + r;
      _Float16* frow = feat + r * CinP;
      if (mode == 0) {
        int nf = nidx[q * K + rr];
        float px = posAll[nf * 3 + 0] - qx;
        float py = posAll[nf * 3 + 1] - qy;
        float pz = posAll[nf * 3 + 2] - qz;
        for (int c = lane; c < CinP; c += 32)
          frow[c] = (_Float16)(c == 0 ? px : (c == 1 ? py : (c == 2 ? pz : 0.f)));
      } else if (mode == 1) {
        int nf = nidx[q * K + rr];
        float px = posAll[nf * 3 + 0] - qx;
        float py = posAll[nf * 3 + 1] - qy;
        float pz = posAll[nf * 3 + 2] - qz;
        const _Float16* xrow = xin + (size_t)nf * xStride;
        for (int c = lane; c < CinP; c += 32) {
          float v;
          if (c < Cx) v = (float)xrow[c];
          else if (c == Cx) v = px;
          else if (c == Cx + 1) v = py;
          else if (c == Cx + 2) v = pz;
          else v = 0.f;
          frow[c] = (_Float16)v;
        }
      } else {
        int xr = q * K + rr;
        const _Float16* xrow = xin + (size_t)xr * xStride;
        for (int c = lane; c < CinP; c += 32) {
          float v;
          if (c < Cx) v = (float)xrow[c];
          else if (posAll != nullptr && c < Cx + 3) v = posAll[(size_t)xr * 3 + (c - Cx)];
          else v = 0.f;
          frow[c] = (_Float16)v;
        }
      }
    }
    __syncthreads();

    // ---- layer 1: feat[16,CinP] @ W1 + b1, ReLU -> h1 ----
    for (int t = wave; t < tiles1; t += NW) {
      int n0 = t << 4;
      v8f acc = {};
      for (int si = 0; si < S1; ++si) {
        v16h a = load_a_lds(feat, CinP, si << 5, col, kh);
        v16h b = load_b_sw(W1, S1, t, si, lane);
        acc = __builtin_amdgcn_wmma_f32_16x16x32_f16(false, a, false, b, (short)0, acc, false, false);
      }
      float bv = B1[n0 + col];
#pragma unroll
      for (int i = 0; i < 8; ++i) {
        float v = acc[i] + bv; v = v > 0.f ? v : 0.f;
        h1[(i + (kh << 3)) * C1 + n0 + col] = (_Float16)v;
      }
    }
    __syncthreads();

    // ---- layer 2: h1 @ W2 + b2, ReLU -> h2 ----
    for (int t = wave; t < tiles2; t += NW) {
      int n0 = t << 4;
      v8f acc = {};
      for (int si = 0; si < S2; ++si) {
        v16h a = load_a_lds(h1, C1, si << 5, col, kh);
        v16h b = load_b_sw(W2, S2, t, si, lane);
        acc = __builtin_amdgcn_wmma_f32_16x16x32_f16(false, a, false, b, (short)0, acc, false, false);
      }
      float bv = B2[n0 + col];
#pragma unroll
      for (int i = 0; i < 8; ++i) {
        float v = acc[i] + bv; v = v > 0.f ? v : 0.f;
        h2[(i + (kh << 3)) * C2 + n0 + col] = (_Float16)v;
      }
    }
    __syncthreads();

    // ---- layer 3: h2 @ W3 + b3 (linear); max-agg in registers or direct store ----
    for (int t = wave, ti = 0; t < tiles3; t += NW, ++ti) {
      int n0 = t << 4;
      v8f acc = {};
      for (int si = 0; si < S3; ++si) {
        v16h a = load_a_lds(h2, C2, si << 5, col, kh);
        v16h b = load_b_sw(W3, S3, t, si, lane);
        acc = __builtin_amdgcn_wmma_f32_16x16x32_f16(false, a, false, b, (short)0, acc, false, false);
      }
      float bv = B3[n0 + col];
      if (doMax) {
        float mx = -3.0e38f;
#pragma unroll
        for (int i = 0; i < 8; ++i) mx = fmaxf(mx, acc[i] + bv);
        mx = fmaxf(mx, __shfl_xor(mx, 16, 32));  // combine row halves (C/D layout)
        accum[ti] = fmaxf(accum[ti], mx);
      } else {
#pragma unroll
        for (int i = 0; i < 8; ++i) {
          int row = (blk << 4) + i + (kh << 3);
          outF32[(size_t)row * outStride + outOff + n0 + col] = acc[i] + bv;
        }
      }
    }
    // no sync needed here: next staging only touches feat (last read before S1)
  }

  if (doMax) {
    for (int t = wave, ti = 0; t < tiles3; t += NW, ++ti) {
      if (lane < 16)
        outF16[(size_t)q * outStride + outOff + (t << 4) + lane] = (_Float16)accum[ti];
    }
  }
}

// ---------------- farthest point sampling: one block per cloud ----------------
__global__ void __launch_bounds__(256)
fps_kernel(const float* __restrict__ pos, int N, int M, int* __restrict__ outIdx) {
  __shared__ float dist[1024];
  __shared__ float redv[256];
  __shared__ int   redi[256];
  __shared__ int   s_cur;
  int b = blockIdx.x, tid = threadIdx.x;
  const float* P = pos + (size_t)b * N * 3;
  for (int i = tid; i < N; i += 256) dist[i] = 3.0e38f;
  int cur = 0;
  __syncthreads();
  for (int m = 0; m < M; ++m) {
    if (tid == 0) outIdx[b * M + m] = b * N + cur;
    float cx = P[cur * 3 + 0], cy = P[cur * 3 + 1], cz = P[cur * 3 + 2];
    float bestv = -1.f; int besti = 0;
    for (int i = tid; i < N; i += 256) {
      float dx = P[i * 3 + 0] - cx, dy = P[i * 3 + 1] - cy, dz = P[i * 3 + 2] - cz;
      float d2 = dx * dx + dy * dy + dz * dz;
      float dm = fminf(dist[i], d2);
      dist[i] = dm;
      if (dm > bestv || (dm == bestv && i < besti)) { bestv = dm; besti = i; }
    }
    redv[tid] = bestv; redi[tid] = besti;
    __syncthreads();
    for (int s = 128; s > 0; s >>= 1) {
      if (tid < s) {
        float ov = redv[tid + s]; int oi = redi[tid + s];
        if (ov > redv[tid] || (ov == redv[tid] && oi < redi[tid])) { redv[tid] = ov; redi[tid] = oi; }
      }
      __syncthreads();
    }
    if (tid == 0) s_cur = redi[0];
    __syncthreads();
    cur = s_cur;
    __syncthreads();
  }
}

// ------------- ball query: first K within radius, pad with first/self -------------
__global__ void __launch_bounds__(256)
ballq_kernel(const float* __restrict__ pos, const float* __restrict__ qpos,
             const int* __restrict__ selfIdx, int N, int M, int Mtot, int K, float r2,
             int* __restrict__ nidx) {
  int q = blockIdx.x * blockDim.x + threadIdx.x;
  if (q >= Mtot) return;
  int b = q / M;
  float qx = qpos[q * 3], qy = qpos[q * 3 + 1], qz = qpos[q * 3 + 2];
  int base = b * N;
  int cnt = 0;
  for (int n = 0; n < N && cnt < K; ++n) {
    int f = base + n;
    float dx = pos[f * 3] - qx, dy = pos[f * 3 + 1] - qy, dz = pos[f * 3 + 2] - qz;
    if (dx * dx + dy * dy + dz * dz <= r2) nidx[(size_t)q * K + (cnt++)] = f;
  }
  int pad = cnt > 0 ? nidx[(size_t)q * K] : selfIdx[q];
  for (; cnt < K; ++cnt) nidx[(size_t)q * K + cnt] = pad;
}

__global__ void __launch_bounds__(256)
gatherq_kernel(const float* __restrict__ pos, const int* __restrict__ idx,
               float* __restrict__ qpos, int total) {
  int q = blockIdx.x * blockDim.x + threadIdx.x;
  if (q < total) {
    int f = idx[q];
    qpos[q * 3 + 0] = pos[f * 3 + 0];
    qpos[q * 3 + 1] = pos[f * 3 + 1];
    qpos[q * 3 + 2] = pos[f * 3 + 2];
  }
}

// pad+convert f32 weights -> f16 in FRAGMENT-SWIZZLED order:
//   Wp[((t*S + si)*32 + lane)*16 + j]  with  col = t*16 + (lane&15), kh = lane>>4,
//   k = si*32 + (j<8 ? kh*8 + j : 16 + kh*8 + (j-8))
// so a B fragment is 32 contiguous bytes per lane (2x b128), coalesced per wave.
__global__ void __launch_bounds__(256)
prepw_kernel(const float* __restrict__ W, const float* __restrict__ b,
             _Float16* __restrict__ Wp, float* __restrict__ bp,
             int Cin, int Cout, int CinP, int CoutP) {
  int S = CinP >> 5;
  int total = CinP * CoutP;
  int i = blockIdx.x * blockDim.x + threadIdx.x;
  if (i < total) {
    int j = i & 15;
    int lane = (i >> 4) & 31;
    int rest = i >> 9;                // t*S + si
    int si = rest % S;
    int t = rest / S;
    int colw = (t << 4) + (lane & 15);
    int kh = lane >> 4;
    int k = (si << 5) + ((j < 8) ? ((kh << 3) + j) : (16 + (kh << 3) + (j - 8)));
    Wp[i] = (k < Cin && colw < Cout) ? (_Float16)W[k * Cout + colw] : (_Float16)0.f;
  }
  if (i < CoutP) bp[i] = (i < Cout) ? b[i] : 0.f;
}

__global__ void logsoftmax_kernel(const float* __restrict__ in, float* __restrict__ out) {
  int b = blockIdx.x;
  if (threadIdx.x == 0) {
    float mx = -3.0e38f;
    for (int j = 0; j < 40; ++j) mx = fmaxf(mx, in[b * 48 + j]);
    float s = 0.f;
    for (int j = 0; j < 40; ++j) s += expf(in[b * 48 + j] - mx);
    float l = logf(s) + mx;
    for (int j = 0; j < 40; ++j) out[b * 40 + j] = in[b * 48 + j] - l;
  }
}

extern "C" void kernel_launch(void* const* d_in, const int* in_sizes, int n_in,
                              void* d_out, int out_size, void* d_ws, size_t ws_size,
                              hipStream_t stream) {
  (void)in_sizes; (void)n_in; (void)out_size; (void)ws_size;
  const float* pos = (const float*)d_in[0];

  struct LD { int cin, cout, cinP, coutP; };
  auto p32 = [](int x) { return (x + 31) & ~31; };
  auto p16 = [](int x) { return (x + 15) & ~15; };
  int dims[8][4] = {
    {3, 32, 32, 128}, {3, 64, 64, 128}, {3, 64, 96, 128},
    {387, 64, 64, 128}, {387, 128, 128, 256}, {387, 128, 128, 256},
    {643, 256, 512, 1024}, {1024, 512, 256, 40}};
  LD L[24];
  for (int m = 0; m < 8; ++m)
    for (int j = 0; j < 3; ++j) {
      int cin = dims[m][j], cout = dims[m][j + 1];
      L[m * 3 + j] = {cin, cout, p32(cin), (j == 2) ? p16(cout) : cout};
    }

  char* base = (char*)d_ws; size_t off = 0;
  auto alloc = [&](size_t bytes) -> void* {
    void* p = base + off; off = (off + bytes + 255) & ~(size_t)255; return p;
  };
  _Float16* Wp[24]; float* Bp[24];
  for (int k = 0; k < 24; ++k) {
    Wp[k] = (_Float16*)alloc((size_t)L[k].cinP * L[k].coutP * 2);
    Bp[k] = (float*)alloc((size_t)L[k].coutP * 4);
  }
  const int B = 16, N = 1024, M1 = 512, M2 = 128;
  int K1[3] = {16, 32, 128};  float R1[3] = {0.1f, 0.2f, 0.4f};
  int K2[3] = {32, 64, 128};  float R2[3] = {0.2f, 0.4f, 0.8f};
  int* fps1 = (int*)alloc((size_t)B * M1 * 4);
  float* qp1 = (float*)alloc((size_t)B * M1 * 3 * 4);
  int* n1[3]; for (int i = 0; i < 3; ++i) n1[i] = (int*)alloc((size_t)B * M1 * K1[i] * 4);
  _Float16* x1 = (_Float16*)alloc((size_t)B * M1 * 384 * 2);
  int* fps2 = (int*)alloc((size_t)B * M2 * 4);
  float* qp2 = (float*)alloc((size_t)B * M2 * 3 * 4);
  int* n2[3]; for (int i = 0; i < 3; ++i) n2[i] = (int*)alloc((size_t)B * M2 * K2[i] * 4);
  _Float16* x2 = (_Float16*)alloc((size_t)B * M2 * 640 * 2);
  _Float16* g  = (_Float16*)alloc((size_t)B * 1024 * 2);
  float* hd = (float*)alloc((size_t)B * 48 * 4);

  // 1) weight prep (f32 -> padded, fragment-swizzled f16)
  for (int k = 0; k < 24; ++k) {
    const float* W  = (const float*)d_in[2 + 2 * k];
    const float* bb = (const float*)d_in[3 + 2 * k];
    int tot = L[k].cinP * L[k].coutP;
    prepw_kernel<<<(tot + 255) / 256, 256, 0, stream>>>(
        W, bb, Wp[k], Bp[k], L[k].cin, L[k].cout, L[k].cinP, L[k].coutP);
  }

  // 2) SA1: fps -> qpos -> ball query -> fused grouped MLPs (out x1 [B,512,384] f16)
  fps_kernel<<<B, 256, 0, stream>>>(pos, N, M1, fps1);
  gatherq_kernel<<<(B * M1 + 255) / 256, 256, 0, stream>>>(pos, fps1, qp1, B * M1);
  for (int i = 0; i < 3; ++i)
    ballq_kernel<<<(B * M1 + 255) / 256, 256, 0, stream>>>(
        pos, qp1, fps1, N, M1, B * M1, K1[i], R1[i] * R1[i], n1[i]);
  int off1[3] = {0, 128, 256};
  for (int i = 0; i < 3; ++i) {
    const LD* l = &L[i * 3];
    grouped_mlp_kernel<<<B * M1, 256, 0, stream>>>(
        0, K1[i], 0, 0, (const _Float16*)nullptr, pos, qp1, n1[i],
        l[0].cinP, l[0].coutP, l[1].coutP, l[2].coutP,
        Wp[i * 3], Bp[i * 3], Wp[i * 3 + 1], Bp[i * 3 + 1], Wp[i * 3 + 2], Bp[i * 3 + 2],
        x1, (float*)nullptr, 384, off1[i], 1);
  }

  // 3) SA2 on (x1, p1=qp1): out x2 [B,128,640] f16
  fps_kernel<<<B, 256, 0, stream>>>(qp1, M1, M2, fps2);
  gatherq_kernel<<<(B * M2 + 255) / 256, 256, 0, stream>>>(qp1, fps2, qp2, B * M2);
  for (int i = 0; i < 3; ++i)
    ballq_kernel<<<(B * M2 + 255) / 256, 256, 0, stream>>>(
        qp1, qp2, fps2, M1, M2, B * M2, K2[i], R2[i] * R2[i], n2[i]);
  int off2[3] = {0, 128, 384};
  for (int i = 0; i < 3; ++i) {
    const LD* l = &L[9 + i * 3];
    grouped_mlp_kernel<<<B * M2, 256, 0, stream>>>(
        1, K2[i], 384, 384, x1, qp1, qp2, n2[i],
        l[0].cinP, l[0].coutP, l[1].coutP, l[2].coutP,
        Wp[9 + i * 3], Bp[9 + i * 3], Wp[9 + i * 3 + 1], Bp[9 + i * 3 + 1],
        Wp[9 + i * 3 + 2], Bp[9 + i * 3 + 2],
        x2, (float*)nullptr, 640, off2[i], 1);
  }

  // 4) SA3 global MLP + max over 128 points -> g [16,1024] f16
  {
    const LD* l = &L[18];
    grouped_mlp_kernel<<<B, 256, 0, stream>>>(
        2, 128, 640, 640, x2, qp2, (const float*)nullptr, (const int*)nullptr,
        l[0].cinP, l[0].coutP, l[1].coutP, l[2].coutP,
        Wp[18], Bp[18], Wp[19], Bp[19], Wp[20], Bp[20],
        g, (float*)nullptr, 1024, 0, 1);
  }

  // 5) head MLP (16 rows, no aggregation) -> hd [16,48] f32, then log_softmax
  {
    const LD* l = &L[21];
    grouped_mlp_kernel<<<1, 256, 0, stream>>>(
        2, 16, 1024, 1024, g, (const float*)nullptr, (const float*)nullptr, (const int*)nullptr,
        l[0].cinP, l[0].coutP, l[1].coutP, l[2].coutP,
        Wp[21], Bp[21], Wp[22], Bp[22], Wp[23], Bp[23],
        (_Float16*)nullptr, hd, 48, 0, 0);
  }
  logsoftmax_kernel<<<B, 64, 0, stream>>>(hd, (float*)d_out);
}
